// PCLSTM_70995809403326
// MI455X (gfx1250) — compile-verified, matrix operand order checked by reading, not run on hardware
//
#include <hip/hip_runtime.h>
#include <math.h>

#define B_SZ 64
#define C_SZ 800
#define T_SZ 512
#define H_SZ 1024
#define IN_SZ 1600   // 2*C
#define G4   4096    // 4*H

#define KC      32   // K-chunk per WMMA stage
#define PAD_ROW 40   // LDS row stride in bf16 elems (32 data + 8 pad) -> 80 B,
                     // bank base = row*20 + seg*4 walks all 64 banks: conflict-free
#define A_ROWS  64
#define B_ROWS  128
#define A_TILE  (A_ROWS * PAD_ROW)   // elems
#define B_TILE  (B_ROWS * PAD_ROW)
#define SMEM_ELEMS (2 * (A_TILE + B_TILE))  // double buffered: 30720 B

typedef __bf16 bf16;
typedef __attribute__((ext_vector_type(16))) __bf16 v16bf;
typedef __attribute__((ext_vector_type(8)))  __bf16 v8bf;
typedef __attribute__((ext_vector_type(8)))  float  v8f;

// ---------------------------------------------------------------------------
// Async global -> LDS copy, 16B per lane (GLOBAL_LOAD_ASYNC_TO_LDS_B128,
// tracked by ASYNCcnt). lds_off is the 32-bit LDS byte address (low 32 bits of
// the generic pointer to __shared__), gaddr a 64-bit global address.
// ---------------------------------------------------------------------------
static __device__ __forceinline__ void async_copy_b128(unsigned lds_off,
                                                       const void* gaddr) {
  asm volatile("global_load_async_to_lds_b128 %0, %1, off"
               : : "v"(lds_off), "v"(gaddr) : "memory");
}
static __device__ __forceinline__ void wait_async0() {
  asm volatile("s_wait_asynccnt 0x0" : : : "memory");
}

// ---------------------------------------------------------------------------
// WMMA bf16 fragment from an LDS tile (row-major, PAD_ROW stride).
// 16-bit A-matrix 16x32 layout (ISA 7.12.2): lanes 0-15 hold row M=lane,
// K = {0..7, 16..23}; lanes 16-31 hold row M=lane-16, K = {8..15, 24..31}.
// B fragments use the same per-lane K pattern with "row" = output column.
// Two ds_load_b128 per lane, 16B aligned.
// ---------------------------------------------------------------------------
static __device__ __forceinline__ v16bf lds_frag(const bf16* __restrict__ tile,
                                                 int row0, int lane) {
  const int r  = lane & 15;
  const int hi = lane >> 4;
  const bf16* p = tile + (row0 + r) * PAD_ROW + hi * 8;
  v8bf lo  = *(const v8bf*)(p);
  v8bf h8  = *(const v8bf*)(p + 16);
  v16bf out;
  #pragma unroll
  for (int j = 0; j < 8; ++j) { out[j] = lo[j]; out[j + 8] = h8[j]; }
  return out;
}

static __device__ __forceinline__ float sigmoidf_fast(float z) {
  return 1.0f / (1.0f + __expf(-z));
}

// ---------------------------------------------------------------------------
// Prep: channel-interleaved bf16 input  xs[t][b][2c]   = x[b][c][t]
//                                       xs[t][b][2c+1] = y[b][c][t-1] (0 at t=0)
// ---------------------------------------------------------------------------
__global__ void k_interleave(const float* __restrict__ x,
                             const float* __restrict__ y,
                             bf16* __restrict__ xs) {
  size_t idx = (size_t)blockIdx.x * blockDim.x + threadIdx.x;  // over T*B*C
  const size_t total = (size_t)T_SZ * B_SZ * C_SZ;
  if (idx >= total) return;
  int c = (int)(idx % C_SZ);
  int b = (int)((idx / C_SZ) % B_SZ);
  int t = (int)(idx / ((size_t)C_SZ * B_SZ));
  size_t src = (size_t)b * C_SZ * T_SZ + (size_t)c * T_SZ + t;
  float xv = x[src];
  float yv = (t > 0) ? y[src - 1] : 0.0f;
  size_t dst = ((size_t)t * B_SZ + b) * IN_SZ + 2 * (size_t)c;
  xs[dst]     = (bf16)xv;
  xs[dst + 1] = (bf16)yv;
}

// Transposed bf16 input weights: WxT[n][k] = W_{g}x[k][h], n = g*H + h
__global__ void k_wxT(const float* __restrict__ Wf, const float* __restrict__ Wi,
                      const float* __restrict__ Wu, const float* __restrict__ Wo,
                      bf16* __restrict__ WxT) {
  size_t idx = (size_t)blockIdx.x * blockDim.x + threadIdx.x;  // 4096*1600
  if (idx >= (size_t)G4 * IN_SZ) return;
  int k = (int)(idx % IN_SZ);
  int n = (int)(idx / IN_SZ);
  int g = n >> 10, h = n & (H_SZ - 1);
  const float* W = (g == 0) ? Wf : (g == 1) ? Wi : (g == 2) ? Wu : Wo;
  WxT[idx] = (bf16)W[(size_t)k * H_SZ + h];
}

// Transposed bf16 recurrent weights: WhT[n][k] = W_{g}h[k][h]
__global__ void k_whT(const float* __restrict__ Wf, const float* __restrict__ Wi,
                      const float* __restrict__ Wu, const float* __restrict__ Wo,
                      bf16* __restrict__ WhT) {
  size_t idx = (size_t)blockIdx.x * blockDim.x + threadIdx.x;  // 4096*1024
  if (idx >= (size_t)G4 * H_SZ) return;
  int k = (int)(idx % H_SZ);
  int n = (int)(idx / H_SZ);
  int g = n >> 10, h = n & (H_SZ - 1);
  const float* W = (g == 0) ? Wf : (g == 1) ? Wi : (g == 2) ? Wu : Wo;
  WhT[idx] = (bf16)W[(size_t)k * H_SZ + h];
}

__global__ void k_bias(const float* __restrict__ bfx, const float* __restrict__ bix,
                       const float* __restrict__ bux, const float* __restrict__ bxo,
                       const float* __restrict__ bfh, const float* __restrict__ bih,
                       const float* __restrict__ buh, const float* __restrict__ bho,
                       float* __restrict__ bx_cat, float* __restrict__ bh_cat) {
  int n = blockIdx.x * blockDim.x + threadIdx.x;  // 4096
  if (n >= G4) return;
  int g = n >> 10, h = n & (H_SZ - 1);
  bx_cat[n] = (g == 0 ? bfx : g == 1 ? bix : g == 2 ? bux : bxo)[h];
  bh_cat[n] = (g == 0 ? bfh : g == 1 ? bih : g == 2 ? buh : bho)[h];
}

// c0 = subject_id @ W_init + b_init   (tiny: 64x87x1024, scalar f32 is fine)
__global__ void k_c0(const float* __restrict__ sid, const float* __restrict__ Winit,
                     const float* __restrict__ binit, float* __restrict__ c_state) {
  int idx = blockIdx.x * blockDim.x + threadIdx.x;  // 65536
  if (idx >= B_SZ * H_SZ) return;
  int h = idx & (H_SZ - 1);
  int b = idx >> 10;
  float acc = binit[h];
  #pragma unroll 1
  for (int k = 0; k < 87; ++k) acc += sid[b * 87 + k] * Winit[k * H_SZ + h];
  c_state[idx] = acc;
}

__global__ void k_zero_h(bf16* __restrict__ h0) {
  int idx = blockIdx.x * blockDim.x + threadIdx.x;
  if (idx < B_SZ * H_SZ) h0[idx] = (bf16)0.0f;
}

// ---------------------------------------------------------------------------
// Big parallel GEMM: gx = xs @ Wx + bx, stored bf16 [T*B, 4H].
// Block = 256 threads = 8 waves laid out 4(M) x 2(N): M-tile 64, N-tile 128.
// Double-buffered async global->LDS staging (ASYNCcnt); fragments via
// ds_load_b128; each wave holds four 16x16 f32 accumulators; K = 1600/32 = 50.
// ---------------------------------------------------------------------------
__global__ __launch_bounds__(256) void k_gx_gemm(const bf16* __restrict__ xs,
                                                 const bf16* __restrict__ WxT,
                                                 const float* __restrict__ bx,
                                                 bf16* __restrict__ gx) {
  __shared__ bf16 smem[SMEM_ELEMS];
  const unsigned lbase = (unsigned)(uintptr_t)(void*)smem;

  const int tid  = threadIdx.x;
  const int lane = tid & 31;
  const int wave = tid >> 5;
  const int wm = wave & 3;
  const int wn = wave >> 2;
  const int m_blk = blockIdx.y * 64;    // row block in [0, T*B)
  const int n_blk = blockIdx.x * 128;   // col block in [0, 4H)

  // LDS byte offsets of the 4 tiles
  const unsigned ldsA[2] = { lbase, lbase + A_TILE * 2u };
  const unsigned ldsB[2] = { lbase + 2u * A_TILE * 2u,
                             lbase + 2u * A_TILE * 2u + B_TILE * 2u };

  auto stage = [&](int buf, int k0) {
    { // A tile: 64 rows x 32 cols = 256 x b128, one per thread
      int row = tid >> 2, seg = tid & 3;
      const bf16* g = xs + (size_t)(m_blk + row) * IN_SZ + k0 + seg * 8;
      async_copy_b128(ldsA[buf] + (unsigned)(row * PAD_ROW + seg * 8) * 2u, g);
    }
    #pragma unroll
    for (int tp = 0; tp < 2; ++tp) {  // B tile: 128 rows x 32 cols = 512 x b128
      int idx = tid + tp * 256;
      int row = idx >> 2, seg = idx & 3;
      const bf16* g = WxT + (size_t)(n_blk + row) * IN_SZ + k0 + seg * 8;
      async_copy_b128(ldsB[buf] + (unsigned)(row * PAD_ROW + seg * 8) * 2u, g);
    }
  };

  const int NK = IN_SZ / KC;  // 50
  v8f acc[4] = {};

  stage(0, 0);
  for (int kc = 0; kc < NK; ++kc) {
    const int buf = kc & 1;
    wait_async0();
    __syncthreads();
    if (kc + 1 < NK) stage(buf ^ 1, (kc + 1) * KC);  // prefetch overlaps compute

    const bf16* As = smem + (buf ? A_TILE : 0);
    const bf16* Bs = smem + 2 * A_TILE + (buf ? B_TILE : 0);
    v16bf a = lds_frag(As, wm * 16, lane);
    #pragma unroll
    for (int nt = 0; nt < 4; ++nt) {
      v16bf b = lds_frag(Bs, wn * 64 + nt * 16, lane);
      acc[nt] = __builtin_amdgcn_wmma_f32_16x16x32_bf16(
          false, a, false, b, (short)0, acc[nt], false, false);
    }
    __syncthreads();
  }

  const int hi = lane >> 4;
  const int cl = lane & 15;
  #pragma unroll
  for (int nt = 0; nt < 4; ++nt) {
    const int n = n_blk + wn * 64 + nt * 16 + cl;
    const float bxi = bx[n];
    #pragma unroll
    for (int j = 0; j < 8; ++j) {
      const int row = m_blk + wm * 16 + hi * 8 + j;
      gx[(size_t)row * G4 + n] = (bf16)(acc[nt][j] + bxi);
    }
  }
}

// ---------------------------------------------------------------------------
// One recurrence step: g = gx[t] + h_prev @ Wh + bh; gates; c,h update.
// Block owns a 32-wide h-slice across ALL FOUR gates (cell update is local).
// Same async-LDS double-buffered pipeline; B tile rows 0..127 map to
// gate = row/32, col = h_blk + row%32 of WhT. Grid = H/32 = 32 blocks.
// ---------------------------------------------------------------------------
__global__ __launch_bounds__(256) void k_step(const bf16* __restrict__ gx_t,
                                              const bf16* __restrict__ h_prev,
                                              bf16* __restrict__ h_next,
                                              const bf16* __restrict__ WhT,
                                              const float* __restrict__ bh,
                                              float* __restrict__ c_state,
                                              float* __restrict__ h_seq,
                                              int t) {
  __shared__ bf16 smem[SMEM_ELEMS];
  const unsigned lbase = (unsigned)(uintptr_t)(void*)smem;

  const int tid  = threadIdx.x;
  const int lane = tid & 31;
  const int wave = tid >> 5;
  const int wm = wave & 3;
  const int wn = wave >> 2;
  const int h_blk = blockIdx.x * 32;

  const unsigned ldsA[2] = { lbase, lbase + A_TILE * 2u };
  const unsigned ldsB[2] = { lbase + 2u * A_TILE * 2u,
                             lbase + 2u * A_TILE * 2u + B_TILE * 2u };

  auto stage = [&](int buf, int k0) {
    { // A tile: h_prev rows 0..63
      int row = tid >> 2, seg = tid & 3;
      const bf16* g = h_prev + (size_t)row * H_SZ + k0 + seg * 8;
      async_copy_b128(ldsA[buf] + (unsigned)(row * PAD_ROW + seg * 8) * 2u, g);
    }
    #pragma unroll
    for (int tp = 0; tp < 2; ++tp) {  // B tile: 4 gates x 32 h-cols
      int idx = tid + tp * 256;
      int row = idx >> 2, seg = idx & 3;
      int grow = (row >> 5) * H_SZ + h_blk + (row & 31);  // WhT row
      const bf16* g = WhT + (size_t)grow * H_SZ + k0 + seg * 8;
      async_copy_b128(ldsB[buf] + (unsigned)(row * PAD_ROW + seg * 8) * 2u, g);
    }
  };

  const int NK = H_SZ / KC;  // 32
  v8f acc[4] = {};

  stage(0, 0);
  for (int kc = 0; kc < NK; ++kc) {
    const int buf = kc & 1;
    wait_async0();
    __syncthreads();
    if (kc + 1 < NK) stage(buf ^ 1, (kc + 1) * KC);

    const bf16* As = smem + (buf ? A_TILE : 0);
    const bf16* Bs = smem + 2 * A_TILE + (buf ? B_TILE : 0);
    v16bf a = lds_frag(As, wm * 16, lane);
    #pragma unroll
    for (int g = 0; g < 4; ++g) {
      v16bf b = lds_frag(Bs, g * 32 + wn * 16, lane);
      acc[g] = __builtin_amdgcn_wmma_f32_16x16x32_bf16(
          false, a, false, b, (short)0, acc[g], false, false);
    }
    __syncthreads();
  }

  const int hi = lane >> 4;
  const int hc = h_blk + wn * 16 + (lane & 15);
  const float bhf = bh[hc];
  const float bhi = bh[H_SZ + hc];
  const float bhu = bh[2 * H_SZ + hc];
  const float bho = bh[3 * H_SZ + hc];

  #pragma unroll
  for (int j = 0; j < 8; ++j) {
    const int bb = wm * 16 + hi * 8 + j;       // batch index
    const size_t gb = (size_t)bb * G4;
    float zf = acc[0][j] + (float)gx_t[gb + hc]            + bhf;
    float zi = acc[1][j] + (float)gx_t[gb + H_SZ + hc]     + bhi;
    float zu = acc[2][j] + (float)gx_t[gb + 2 * H_SZ + hc] + bhu;
    float zo = acc[3][j] + (float)gx_t[gb + 3 * H_SZ + hc] + bho;
    float f = sigmoidf_fast(zf);
    float i = sigmoidf_fast(zi);
    float u = tanhf(zu);
    float o = sigmoidf_fast(zo);
    const size_t ci = (size_t)bb * H_SZ + hc;
    float c = c_state[ci];
    c = c * f + i * u;
    c_state[ci] = c;
    float h = o * tanhf(c);
    h_seq[(size_t)bb * H_SZ * T_SZ + (size_t)hc * T_SZ + t] = h;  // [B,H,T]
    h_next[ci] = (bf16)h;
  }
}

// ---------------------------------------------------------------------------
extern "C" void kernel_launch(void* const* d_in, const int* in_sizes, int n_in,
                              void* d_out, int out_size, void* d_ws, size_t ws_size,
                              hipStream_t stream) {
  const float* x     = (const float*)d_in[0];
  const float* y     = (const float*)d_in[1];
  const float* sid   = (const float*)d_in[2];
  const float* Wfx   = (const float*)d_in[3];
  const float* bfx   = (const float*)d_in[4];
  const float* Wfh   = (const float*)d_in[5];
  const float* bfh   = (const float*)d_in[6];
  const float* Wix   = (const float*)d_in[7];
  const float* bix   = (const float*)d_in[8];
  const float* Wih   = (const float*)d_in[9];
  const float* bih   = (const float*)d_in[10];
  const float* Wux   = (const float*)d_in[11];
  const float* bux   = (const float*)d_in[12];
  const float* Wuh   = (const float*)d_in[13];
  const float* buh   = (const float*)d_in[14];
  const float* Wox   = (const float*)d_in[15];
  const float* bxo   = (const float*)d_in[16];
  const float* Woh   = (const float*)d_in[17];
  const float* bho   = (const float*)d_in[18];
  const float* Winit = (const float*)d_in[19];
  const float* binit = (const float*)d_in[20];
  (void)in_sizes; (void)n_in; (void)out_size; (void)ws_size;

  char* ws = (char*)d_ws;
  size_t off = 0;
  auto walloc = [&](size_t bytes) -> void* {
    void* p = ws + off;
    off = (off + bytes + 255) & ~(size_t)255;
    return p;
  };
  bf16*  xs   = (bf16*) walloc((size_t)T_SZ * B_SZ * IN_SZ * sizeof(bf16));
  bf16*  gx   = (bf16*) walloc((size_t)T_SZ * B_SZ * G4 * sizeof(bf16));
  bf16*  WxT  = (bf16*) walloc((size_t)G4 * IN_SZ * sizeof(bf16));
  bf16*  WhT  = (bf16*) walloc((size_t)G4 * H_SZ * sizeof(bf16));
  float* bx   = (float*)walloc((size_t)G4 * sizeof(float));
  float* bh   = (float*)walloc((size_t)G4 * sizeof(float));
  float* cst  = (float*)walloc((size_t)B_SZ * H_SZ * sizeof(float));
  bf16*  hb0  = (bf16*) walloc((size_t)B_SZ * H_SZ * sizeof(bf16));
  bf16*  hb1  = (bf16*) walloc((size_t)B_SZ * H_SZ * sizeof(bf16));

  float* h_seq = (float*)d_out;
  float* c_fin = h_seq + (size_t)B_SZ * H_SZ * T_SZ;

  // --- prep ---
  {
    size_t n = (size_t)T_SZ * B_SZ * C_SZ;
    k_interleave<<<(unsigned)((n + 255) / 256), 256, 0, stream>>>(x, y, xs);
  }
  {
    size_t n = (size_t)G4 * IN_SZ;
    k_wxT<<<(unsigned)((n + 255) / 256), 256, 0, stream>>>(Wfx, Wix, Wux, Wox, WxT);
  }
  {
    size_t n = (size_t)G4 * H_SZ;
    k_whT<<<(unsigned)((n + 255) / 256), 256, 0, stream>>>(Wfh, Wih, Wuh, Woh, WhT);
  }
  k_bias<<<G4 / 256, 256, 0, stream>>>(bfx, bix, bux, bxo, bfh, bih, buh, bho, bx, bh);
  k_c0<<<(B_SZ * H_SZ) / 256, 256, 0, stream>>>(sid, Winit, binit, cst);
  k_zero_h<<<(B_SZ * H_SZ) / 256, 256, 0, stream>>>(hb0);

  // --- input-side fused gate GEMM over all timesteps ---
  {
    dim3 grid(G4 / 128, (T_SZ * B_SZ) / 64);  // (32, 512)
    k_gx_gemm<<<grid, 256, 0, stream>>>(xs, WxT, bx, gx);
  }

  // --- sequential recurrence ---
  for (int t = 0; t < T_SZ; ++t) {
    const bf16* hp = (t & 1) ? hb1 : hb0;
    bf16*       hn = (t & 1) ? hb0 : hb1;
    k_step<<<H_SZ / 32, 256, 0, stream>>>(gx + (size_t)t * B_SZ * G4,
                                          hp, hn, WhT, bh, cst, h_seq, t);
  }

  hipMemcpyAsync(c_fin, cst, (size_t)B_SZ * H_SZ * sizeof(float),
                 hipMemcpyDeviceToDevice, stream);
}
